// criss_cross_attention_Affinity_14413910245606
// MI455X (gfx1250) — compile-verified, matrix operand order checked by reading, not run on hardware
//
#include <hip/hip_runtime.h>

typedef __attribute__((ext_vector_type(16))) __bf16 v16bf;
typedef __attribute__((ext_vector_type(8)))  __bf16 bf16x8;
typedef __attribute__((ext_vector_type(4)))  __bf16 bf16x4;
typedef __attribute__((ext_vector_type(8)))  float  v8f;

#define DIM_B 8
#define DIM_H 128
#define DIM_W 128
#define DIM_C 64
#define LDS_STRIDE 72   // bf16 per row: 64 padded to 72 (144 B, 16B-aligned, conflict-free b128 reads)

// Load a 16x32-bf16 A/B fragment for v_wmma_f32_16x16x32_bf16 from LDS.
// ISA layout (16-bit A 16x32): lanes 0-15 hold K = {+0..7, +16..23},
// lanes 16-31 hold K = {+8..15, +24..31}; row = lane & 15 within the tile.
// Loading B-fragments with the same code from rows of X yields B = X^T (symmetric GEMM).
__device__ __forceinline__ v16bf cc_load_frag(const __bf16* Xs, int row, int kk, int hoff) {
    const __bf16* p = Xs + row * LDS_STRIDE + kk * 32 + hoff;
    bf16x8 lo = *(const bf16x8*)(p);        // 16 B -> ds_load_b128
    bf16x8 hi = *(const bf16x8*)(p + 16);   // 16 B -> ds_load_b128
    v16bf a;
#pragma unroll
    for (int e = 0; e < 8; ++e) { a[e] = lo[e]; a[e + 8] = hi[e]; }
    return a;
}

template <bool COL_MODE>
__global__ __launch_bounds__(256)
void cc_affinity_kernel(const float* __restrict__ x, float* __restrict__ out) {
    __shared__ __align__(16) __bf16 Xs[DIM_H * LDS_STRIDE];

    const int blk = blockIdx.x;      // b*128 + (i for row mode, j for col mode)
    const int b   = blk >> 7;
    const int ij  = blk & 127;
    const int tid = threadIdx.x;

    // ---- Stage 128x64 fp32 slab -> bf16 LDS (256 threads x 8 float4 each) ----
    const float4* __restrict__ xv = (const float4*)x;
#pragma unroll
    for (int it = 0; it < 8; ++it) {
        const int lin = it * 256 + tid;     // 0..2047 float4 slots
        const int row = lin >> 4;           // 0..127
        const int c4  = lin & 15;           // float4 within the 64-wide channel dim
        size_t gidx;
        if (COL_MODE) {
            gidx = ((size_t)((b * 128 + row) * 128 + ij) << 4) + (size_t)c4; // x[b,row,ij,:]
        } else {
            gidx = ((size_t)((b * 128 + ij) * 128 + row) << 4) + (size_t)c4; // x[b,ij,row,:]
        }
        const float4 v = xv[gidx];
        bf16x4 h;
        h[0] = (__bf16)v.x; h[1] = (__bf16)v.y; h[2] = (__bf16)v.z; h[3] = (__bf16)v.w;
        *(bf16x4*)&Xs[row * LDS_STRIDE + c4 * 4] = h;
    }
    __syncthreads();

    // ---- 8 waves x 8 tile-columns of 16x16 tiles; K=64 -> 2 WMMAs per tile ----
    // readfirstlane: prove wave-uniformity so tile dispatch becomes scalar branches.
    const int wave = __builtin_amdgcn_readfirstlane(tid >> 5);   // tile row mt, SGPR
    const int lane = tid & 31;
    const int ln   = lane & 15;
    const int hi8  = (lane >> 4) & 1;     // lane half: K sub-block select / M+8
    const int hoff = hi8 * 8;

    const v16bf a0 = cc_load_frag(Xs, wave * 16 + ln, 0, hoff);
    const v16bf a1 = cc_load_frag(Xs, wave * 16 + ln, 1, hoff);

    // Per-lane output base; per-r stride folds into the store's 24-bit imm offset.
    float* po;
    if (COL_MODE) {
        // out[b, wave*16+hi8*8+r, ij, 127 + N] ; per-r stride = 128*255 floats
        po = out + (size_t)((b * 128 + wave * 16 + hi8 * 8) * 128 + ij) * 255 + 127;
    } else {
        // out[b, ij, wave*16+hi8*8+r, idx(N)] ; per-r stride = 255 floats
        po = out + (size_t)(blk * 128 + wave * 16 + hi8 * 8) * 255;
    }

#pragma unroll
    for (int nt = 0; nt < 8; ++nt) {
        const v16bf b0 = cc_load_frag(Xs, nt * 16 + ln, 0, hoff);
        const v16bf b1 = cc_load_frag(Xs, nt * 16 + ln, 1, hoff);

        v8f c = {};
        c = __builtin_amdgcn_wmma_f32_16x16x32_bf16(false, a0, false, b0, (short)0, c, false, false);
        c = __builtin_amdgcn_wmma_f32_16x16x32_bf16(false, a1, false, b1, (short)0, c, false, false);

        const int N0 = nt * 16 + ln;          // global column within the 128-wide GEMM
        if (COL_MODE) {
#pragma unroll
            for (int r = 0; r < 8; ++r) po[(size_t)r * (128 * 255) + N0] = c[r];
        } else if (nt != wave) {
            // Off-diagonal tile: self-removal shift is tile-uniform (scalar branch),
            // stores are straight-line.
            const int idx = N0 - (nt > wave ? 1 : 0);
#pragma unroll
            for (int r = 0; r < 8; ++r) po[(size_t)r * 255 + idx] = c[r];
        } else {
            // Diagonal tile: branchless. The self entry (k==j) is redirected to
            // slot 127 of its own row — that slot belongs to the column part and
            // is fully overwritten by the COL_MODE kernel launched afterwards.
#pragma unroll
            for (int r = 0; r < 8; ++r) {
                const int Ml  = hi8 * 8 + r;                    // local row within tile
                const int idx = (ln == Ml) ? 127 : (N0 - (ln > Ml ? 1 : 0));
                po[(size_t)r * 255 + idx] = c[r];
            }
        }
    }
}

extern "C" void kernel_launch(void* const* d_in, const int* in_sizes, int n_in,
                              void* d_out, int out_size, void* d_ws, size_t ws_size,
                              hipStream_t stream) {
    (void)in_sizes; (void)n_in; (void)out_size; (void)d_ws; (void)ws_size;
    const float* x = (const float*)d_in[0];
    float* out = (float*)d_out;

    // Row affinities first (its diagonal dump writes land in the column region),
    // then column affinities overwrite the full [127..254] range of every row.
    cc_affinity_kernel<false><<<DIM_B * DIM_H, 256, 0, stream>>>(x, out);
    cc_affinity_kernel<true ><<<DIM_B * DIM_W, 256, 0, stream>>>(x, out);
}
